// ContextPulseAttention_1992864825630
// MI455X (gfx1250) — matrix operand order repaired and checked by reading, not compiled, not run on hardware
//
#include <hip/hip_runtime.h>
#include <hip/hip_bf16.h>

// ---------------------------------------------------------------------------
// ContextPulseAttention for MI455X (gfx1250), wave32 + WMMA f16 (f32 accum).
//   1) cast x -> f16, build concat weights (f16), build S^T (f16), later v^T
//   2) proj GEMM (WMMA): [q|k|v|g] = x @ Wcat^T   (g gets sigmoid+bias)
//   3) serial chunk scan (fp32, LDS-resident history)  -> Qc
//   4) per-token norm + RoPE -> Rq, Rk (f16); v -> vhT (f16, transposed)
//   5) wedge GEMM (WMMA): Qg = Rq + Rq@S ; Kg = Rk + Rk@S   (B = S^T, NK form)
//   6) score GEMM (WMMA): attn = Qg @ Kg^T   (causal tiles skipped)
//   7) row softmax with sink column -> probs (f16, masked = 0), psink
//   8) PV GEMM (WMMA): out = probs @ v + psink * v_null     (B = v^T, NK form)
// GEMM core: 128x128 block tile, 8 waves, 32Mx64N per wave (8 WMMA/k-step).
// Tile staging via Tensor Data Mover (double-buffered, hardware LDS padding)
// when the builtin is available; register-staged b128 copies otherwise.
// ---------------------------------------------------------------------------

typedef __attribute__((ext_vector_type(16))) _Float16 v16h;
typedef __attribute__((ext_vector_type(8)))  _Float16 h8;
typedef __attribute__((ext_vector_type(8)))  float    v8f;

#define BB   8
#define TT   2048
#define DD   512
#define CSZ  64
#define NC   32
#define NROW (BB*TT)           // 16384

#if __has_builtin(__builtin_amdgcn_tensor_load_to_lds) && \
    __has_builtin(__builtin_amdgcn_s_wait_tensorcnt)
#define CPA_TDM 1
#endif

#ifdef CPA_TDM
#define CPA_TILE_ELEMS (2 * 128 * 40)   // double-buffered
#else
#define CPA_TILE_ELEMS (128 * 40)
#endif

// ============================ small converters =============================

__global__ void cpa_cast_f16(const float* __restrict__ src,
                             _Float16* __restrict__ dst, int n) {
  int i = blockIdx.x * 256 + threadIdx.x;
  if (i < n) dst[i] = (_Float16)src[i];
}

__global__ void cpa_build_wcat(const float* __restrict__ Wq,
                               const float* __restrict__ Wk,
                               const float* __restrict__ Wv,
                               const float* __restrict__ Wg,
                               _Float16* __restrict__ W) {
  int i = blockIdx.x * 256 + threadIdx.x;
  if (i >= 2048 * 512) return;
  int r = i >> 9, c = i & 511;
  const float* src = (r < 512) ? Wq : (r < 1024) ? Wk : (r < 1536) ? Wv : Wg;
  W[i] = (_Float16)src[(r & 511) * 512 + c];
}

// ShT[e][d] = S[d][e] = A[d][e] - A[e][d]  (N x K layout for the wedge GEMM)
__global__ void cpa_build_st(const float* __restrict__ A,
                             _Float16* __restrict__ ShT) {
  int i = blockIdx.x * 256 + threadIdx.x;
  if (i >= 512 * 512) return;
  int e = i >> 9, d = i & 511;
  ShT[i] = (_Float16)(A[d * 512 + e] - A[e * 512 + d]);
}

// ============================ WMMA fragment helpers ========================
// ISA 7.12.2 layouts (wave32):
//  A 16x32 f16 : lane<16 -> row=lane,    elems 0..7 = K0..7,  8..15 = K16..23
//                lane>=16 -> row=lane-16, elems 0..7 = K8..15, 8..15 = K24..31
//  B 32x16 f16 : lane<16 -> col=lane,    elems 0..7 = K0..7,  8..15 = K16..23
//                lane>=16 -> col=lane-16, elems 0..7 = K8..15, 8..15 = K24..31
//  C/D 16x16 f32: lane<16 -> N=lane, M = vgpr (0..7); lane>=16 -> M = vgpr+8

__device__ __forceinline__ v16h cpa_frag_a(const _Float16* Al, int ld, int mbase) {
  int lane = threadIdx.x & 31;
  int m = mbase + (lane & 15);
  int koff = (lane < 16) ? 0 : 8;
  h8 lo = *(const h8*)(Al + m * ld + koff);
  h8 hi = *(const h8*)(Al + m * ld + koff + 16);
  v16h a;
#pragma unroll
  for (int e = 0; e < 8; ++e) { a[e] = lo[e]; a[8 + e] = hi[e]; }
  return a;
}

__device__ __forceinline__ v16h cpa_frag_b_nk(const _Float16* Bl, int ld, int nbase) {
  int lane = threadIdx.x & 31;
  int n = nbase + (lane & 15);
  int kb = (lane < 16) ? 0 : 8;
  h8 lo = *(const h8*)(Bl + n * ld + kb);
  h8 hi = *(const h8*)(Bl + n * ld + kb + 16);
  v16h b;
#pragma unroll
  for (int e = 0; e < 8; ++e) { b[e] = lo[e]; b[8 + e] = hi[e]; }
  return b;
}

#define CPA_WMMA(acc, a, b) \
  __builtin_amdgcn_wmma_f32_16x16x32_f16(false, (a), false, (b), (short)0, (acc), false, false)

// ======================= TDM tile loader (gfx1250) =========================
#ifdef CPA_TDM
typedef unsigned int cpa_u32x4 __attribute__((ext_vector_type(4)));
typedef int cpa_i32x8 __attribute__((ext_vector_type(8)));
typedef int cpa_i32x4 __attribute__((ext_vector_type(4)));

// DMA one 128-row x 32-half tile (row stride = lda halves) into LDS with
// hardware padding to a 40-half row pitch (pad_interval=16 DW, pad_amount=4 DW).
__device__ __forceinline__ void cpa_tdm_load_tile(const _Float16* g, unsigned lda,
                                                  _Float16* lds) {
  unsigned long long ga = (unsigned long long)(uintptr_t)g;
  // generic pointer to __shared__: low 32 bits carry the LDS byte offset
  unsigned ldsoff = (unsigned)(uintptr_t)lds;
  cpa_u32x4 g0;
  g0[0] = 1u;                                        // count=1 (valid D#)
  g0[1] = ldsoff;                                    // lds_addr
  g0[2] = (unsigned)(ga & 0xFFFFFFFFull);            // global_addr[31:0]
  g0[3] = ((unsigned)(ga >> 32) & 0x01FFFFFFu)       // global_addr[56:32]
          | 0x80000000u;                             // type = 2 ("image")
  cpa_i32x8 g1;
  g1[0] = (1 << 16)                                  // data_size = 2 bytes
        | (1 << 20)                                  // pad_enable
        | (3 << 22)                                  // pad_interval: 16 DWORDs
        | (3 << 25);                                 // pad_amount:   4 DWORDs
  g1[1] = (int)((32u & 0xFFFFu) << 16);              // tensor_dim0 = 32 (lo)
  g1[2] = (int)((128u & 0xFFFFu) << 16);             // dim0 hi=0 | tensor_dim1=128
  g1[3] = (int)(32u << 16);                          // dim1 hi=0 | tile_dim0 = 32
  g1[4] = 128;                                       // tile_dim1 = 128, tile_dim2 = 0
  g1[5] = (int)lda;                                  // tensor_dim0_stride (lo)
  g1[6] = 0;                                         // stride hi | dim1_stride lo
  g1[7] = 0;
  cpa_i32x4 z4 = {0, 0, 0, 0};
  cpa_i32x8 z8 = {0, 0, 0, 0, 0, 0, 0, 0};
  // clang-23 / therock-10.0 lane: 6-arg form
  __builtin_amdgcn_tensor_load_to_lds(g0, g1, z4, z4, z8, 0);
}
#endif

// =================== shared 128x128 WMMA main loop =========================
// Block: 256 threads = 8 waves; wave (wm,wn) in 4x2; per wave 32M x 64N:
//   acc[2][4] (16x16 f32 tiles); per k-step: 2 A-frags + 4 B-frags -> 8 WMMA.
// Both A (M x K) and B (N x K) are row-major with leading dims lda/ldb.

__device__ __forceinline__ void cpa_mma_loop(
    const _Float16* __restrict__ Aglob, size_t lda,
    const _Float16* __restrict__ Bglob, size_t ldb,
    int m0, int n0, int kend,
    _Float16* As, _Float16* Bs, v8f acc[2][4]) {
  int tid = threadIdx.x, wid = tid >> 5;
  int wm = wid >> 1, wn = wid & 1;
#ifdef CPA_TDM
  bool issuer = (wid == 0);
  int nsteps = kend >> 5;
  if (issuer) {   // prologue: fill buffer 0
    cpa_tdm_load_tile(Aglob + (size_t)m0 * lda, (unsigned)lda, As);
    cpa_tdm_load_tile(Bglob + (size_t)n0 * ldb, (unsigned)ldb, Bs);
  }
  for (int s = 0; s < nsteps; ++s) {
    int cur = s & 1, nxt = cur ^ 1;
    _Float16* Ac = As + cur * (128 * 40);
    _Float16* Bc = Bs + cur * (128 * 40);
    if (issuer) {
      if (s + 1 < nsteps) {       // kick next tile pair, then wait for current
        int k1 = (s + 1) << 5;
        cpa_tdm_load_tile(Aglob + (size_t)m0 * lda + k1, (unsigned)lda,
                          As + nxt * (128 * 40));
        cpa_tdm_load_tile(Bglob + (size_t)n0 * ldb + k1, (unsigned)ldb,
                          Bs + nxt * (128 * 40));
        __builtin_amdgcn_s_wait_tensorcnt(2);   // in-order: current pair done
      } else {
        __builtin_amdgcn_s_wait_tensorcnt(0);
      }
    }
    __syncthreads();
    v16h a0 = cpa_frag_a(Ac, 40, wm * 32);
    v16h a1 = cpa_frag_a(Ac, 40, wm * 32 + 16);
#pragma unroll
    for (int t = 0; t < 4; ++t) {
      v16h b = cpa_frag_b_nk(Bc, 40, wn * 64 + t * 16);
      acc[0][t] = CPA_WMMA(acc[0][t], a0, b);
      acc[1][t] = CPA_WMMA(acc[1][t], a1, b);
    }
    __syncthreads();
  }
#else
  for (int k0 = 0; k0 < kend; k0 += 32) {
    h8 ra[2], rb[2];
#pragma unroll
    for (int j = 0; j < 2; ++j) {   // gather first: loads can clause together
      int idx = (tid + j * 256) * 8, r = idx >> 5, c = idx & 31;
      ra[j] = *(const h8*)(Aglob + (size_t)(m0 + r) * lda + k0 + c);
      rb[j] = *(const h8*)(Bglob + (size_t)(n0 + r) * ldb + k0 + c);
    }
#pragma unroll
    for (int j = 0; j < 2; ++j) {
      int idx = (tid + j * 256) * 8, r = idx >> 5, c = idx & 31;
      *(h8*)&As[r * 40 + c] = ra[j];
      *(h8*)&Bs[r * 40 + c] = rb[j];
    }
    if (k0 + 32 < kend)
      __builtin_prefetch(Aglob + (size_t)(m0 + (tid & 127)) * lda + k0 + 32, 0, 0);
    __syncthreads();
    v16h a0 = cpa_frag_a(As, 40, wm * 32);
    v16h a1 = cpa_frag_a(As, 40, wm * 32 + 16);
#pragma unroll
    for (int t = 0; t < 4; ++t) {
      v16h b = cpa_frag_b_nk(Bs, 40, wn * 64 + t * 16);
      acc[0][t] = CPA_WMMA(acc[0][t], a0, b);
      acc[1][t] = CPA_WMMA(acc[1][t], a1, b);
    }
    __syncthreads();
  }
#endif
}

// epilogue index helpers: row/col of acc[s][t] element v for this lane
#define CPA_EPILOGUE_BEGIN                                              \
  int lane = threadIdx.x & 31, nl = lane & 15, mb = (lane < 16) ? 0 : 8; \
  int wid_ = threadIdx.x >> 5, wm_ = wid_ >> 1, wn_ = wid_ & 1;          \
  _Pragma("unroll") for (int s = 0; s < 2; ++s)                          \
  _Pragma("unroll") for (int t = 0; t < 4; ++t)                          \
  _Pragma("unroll") for (int v = 0; v < 8; ++v) {                        \
    int row = m0 + wm_ * 32 + s * 16 + mb + v;                           \
    int col = n0 + wn_ * 64 + t * 16 + nl;                               \
    float val = acc[s][t][v];
#define CPA_EPILOGUE_END }

// ======================= 2) projection GEMM (WMMA) =========================

__global__ __launch_bounds__(256) void cpa_proj_gemm(
    const _Float16* __restrict__ X, const _Float16* __restrict__ W,
    const float* __restrict__ bg, float* __restrict__ qb, float* __restrict__ kb,
    float* __restrict__ vb, float* __restrict__ gb) {
  __shared__ _Float16 As[CPA_TILE_ELEMS];
  __shared__ _Float16 Bs[CPA_TILE_ELEMS];
  int m0 = blockIdx.x * 128, n0 = blockIdx.y * 128;
  v8f acc[2][4] = {};
  cpa_mma_loop(X, 512, W, 512, m0, n0, 512, As, Bs, acc);

  CPA_EPILOGUE_BEGIN
    if (col < 512)       qb[(size_t)row * 512 + col] = val;
    else if (col < 1024) kb[(size_t)row * 512 + col - 512] = val;
    else if (col < 1536) vb[(size_t)row * 512 + col - 1024] = val;
    else { float z = val + bg[col - 1536];
           gb[(size_t)row * 512 + col - 1536] = 1.0f / (1.0f + __expf(-z)); }
  CPA_EPILOGUE_END
}

// ============================ 3) serial chunk scan =========================

__global__ __launch_bounds__(512) void cpa_scan(
    const float* __restrict__ q, const float* __restrict__ k,
    const float* __restrict__ g, float* __restrict__ Qc) {
  __shared__ float hist[NC][DD];   // 64 KB
  __shared__ float kc[CSZ][DD];    // 128 KB
  __shared__ float w[CSZ][NC];     // 8 KB
  __shared__ float last[DD];       // 2 KB
  int b = blockIdx.x, tid = threadIdx.x;
  const float scale = rsqrtf((float)DD);

  for (int n = 0; n < NC; ++n) hist[n][tid] = 0.0f;
  last[tid] = 0.0f;
  __syncthreads();

  for (int i = 0; i < NC; ++i) {
    int rowbase = (b * NC + i) * CSZ;
    for (int c = 0; c < CSZ; ++c)
      kc[c][tid] = k[(size_t)(rowbase + c) * DD + tid];
    __syncthreads();

    for (int p = tid; p < CSZ * NC; p += 512) {
      int c = p >> 5, n = p & 31;
      float s = -1e30f;
      if (n < i) {
        s = 0.0f;
        for (int d = 0; d < DD; ++d) s += kc[c][d] * hist[n][d];
        s *= scale;
      }
      w[c][n] = s;
    }
    __syncthreads();

    if (tid < CSZ) {   // per-row masked softmax over history slots
      float m = -1e30f;
      for (int n = 0; n < i; ++n) m = fmaxf(m, w[tid][n]);
      float den = 0.0f;
      for (int n = 0; n < i; ++n) { float e = __expf(w[tid][n] - m); w[tid][n] = e; den += e; }
      float inv = (i > 0) ? 1.0f / den : 0.0f;
      for (int n = 0; n < i; ++n)  w[tid][n] *= inv;
      for (int n = i; n < NC; ++n) w[tid][n] = 0.0f;
    }
    __syncthreads();

    int d = tid;
    float run = 0.0f, lastd = last[d], outc = 0.0f;
    for (int c = 0; c < CSZ; ++c) {
      size_t r = (size_t)(rowbase + c) * DD + d;
      run += g[r] * q[r];
      float qretr = 0.0f;
      for (int n = 0; n < i; ++n) qretr += w[c][n] * hist[n][d];
      outc = run + ((i > 0) ? (lastd + qretr) : 0.0f);
      Qc[r] = outc;
    }
    hist[i][d] = outc;
    last[d] = outc;
    __syncthreads();
  }
}

// ======================= 4) norm + RoPE + v->f16^T =========================

__global__ __launch_bounds__(512) void cpa_norm_rope(
    const float* __restrict__ Qc, const float* __restrict__ k,
    const float* __restrict__ v, _Float16* __restrict__ Rq,
    _Float16* __restrict__ Rk, _Float16* __restrict__ vhT) {
  __shared__ float sQ[DD], sK[DD], red[DD];
  int row = blockIdx.x, tid = threadIdx.x;
  size_t base = (size_t)row * DD;
  float qv = Qc[base + tid], kv = k[base + tid];

  auto blk_sum = [&](float x) -> float {
    red[tid] = x; __syncthreads();
    for (int s = 256; s > 0; s >>= 1) { if (tid < s) red[tid] += red[tid + s]; __syncthreads(); }
    float r = red[0]; __syncthreads(); return r;
  };

  float mQ = blk_sum(qv) * (1.0f / DD);
  float mK = blk_sum(kv) * (1.0f / DD);
  float dq = qv - mQ, dk = kv - mK;
  float stdQ = sqrtf(blk_sum(dq * dq) / (DD - 1));
  float stdK = sqrtf(blk_sum(dk * dk) / (DD - 1));
  float mean_s = 0.5f * (mQ + mK);
  float std_s  = 0.5f * (stdQ + stdK) + 1e-6f;
  float inv = 1.0f / std_s;
  sQ[tid] = (qv - mean_s) * inv;
  sK[tid] = (kv - mean_s) * inv;
  vhT[(size_t)tid * NROW + row] = (_Float16)v[base + tid];   // transposed store
  __syncthreads();

  int t = row & (TT - 1);
  if (tid < 256) {
    float invf = __expf(-((float)(2 * tid) / (float)DD) * 9.210340371976184f);
    float ang = (float)t * invf, s, c;
    __sincosf(ang, &s, &c);
    float q1 = sQ[2 * tid], q2 = sQ[2 * tid + 1];
    Rq[base + tid]       = (_Float16)(q1 * c - q2 * s);
    Rq[base + 256 + tid] = (_Float16)(q1 * s + q2 * c);
    float k1 = sK[2 * tid], k2 = sK[2 * tid + 1];
    Rk[base + tid]       = (_Float16)(k1 * c - k2 * s);
    Rk[base + 256 + tid] = (_Float16)(k1 * s + k2 * c);
  }
}

// ===================== 5) wedge GEMM (WMMA): R + R@S =======================

__global__ __launch_bounds__(256) void cpa_wedge_gemm(
    const _Float16* __restrict__ Rq, const _Float16* __restrict__ Rk,
    const _Float16* __restrict__ ShT, _Float16* __restrict__ Qg,
    _Float16* __restrict__ Kg) {
  const _Float16* Am = blockIdx.z ? Rk : Rq;
  _Float16* D = blockIdx.z ? Kg : Qg;
  __shared__ _Float16 As[CPA_TILE_ELEMS];
  __shared__ _Float16 Bs[CPA_TILE_ELEMS];
  int m0 = blockIdx.x * 128, n0 = blockIdx.y * 128;
  v8f acc[2][4] = {};
  cpa_mma_loop(Am, 512, ShT, 512, m0, n0, 512, As, Bs, acc);

  CPA_EPILOGUE_BEGIN
    float o = val + (float)Am[(size_t)row * 512 + col];
    D[(size_t)row * 512 + col] = (_Float16)o;
  CPA_EPILOGUE_END
}

// ================== 6) score GEMM (WMMA): Qg @ Kg^T ========================

__global__ __launch_bounds__(256) void cpa_score_gemm(
    const _Float16* __restrict__ Qg, const _Float16* __restrict__ Kg,
    float* __restrict__ attn) {
  int b = blockIdx.z;
  int m0 = blockIdx.x * 128, n0 = blockIdx.y * 128;
  if (n0 >= m0 + 128) return;  // fully-masked causal tile
  const _Float16* Ab = Qg + (size_t)b * TT * 512;
  const _Float16* Bb = Kg + (size_t)b * TT * 512;
  __shared__ _Float16 As[CPA_TILE_ELEMS];
  __shared__ _Float16 Bs[CPA_TILE_ELEMS];
  v8f acc[2][4] = {};
  cpa_mma_loop(Ab, 512, Bb, 512, m0, n0, 512, As, Bs, acc);

  CPA_EPILOGUE_BEGIN
    attn[((size_t)b * TT + row) * TT + col] = val;
  CPA_EPILOGUE_END
}

// ================= 7) causal softmax with sink column ======================

__global__ __launch_bounds__(256) void cpa_softmax(
    const float* __restrict__ attn, const float* __restrict__ sinkp,
    _Float16* __restrict__ probs, float* __restrict__ psink) {
  __shared__ float red[256];
  int row = blockIdx.x, tid = threadIdx.x;
  int tr = row & (TT - 1);
  const float scale = rsqrtf((float)DD);
  const float* arow = attn + (size_t)row * TT;
  _Float16* prow = probs + (size_t)row * TT;
  float sink = sinkp[0];

  float m = sink;
  for (int s = tid; s <= tr; s += 256) m = fmaxf(m, arow[s] * scale);
  red[tid] = m; __syncthreads();
  for (int st = 128; st > 0; st >>= 1) { if (tid < st) red[tid] = fmaxf(red[tid], red[tid + st]); __syncthreads(); }
  m = red[0]; __syncthreads();

  float sum = 0.0f;
  for (int s = tid; s <= tr; s += 256) sum += __expf(arow[s] * scale - m);
  red[tid] = sum; __syncthreads();
  for (int st = 128; st > 0; st >>= 1) { if (tid < st) red[tid] += red[tid + st]; __syncthreads(); }
  float total = red[0] + __expf(sink - m);
  float inv = 1.0f / total;

  for (int s = tid; s < TT; s += 256)
    prow[s] = (_Float16)((s <= tr) ? __expf(arow[s] * scale - m) * inv : 0.0f);
  if (tid == 0) psink[row] = __expf(sink - m) * inv;
}

// ===== 8) PV GEMM (WMMA): out = probs @ v + psink * v_null (per batch) =====

__global__ __launch_bounds__(256) void cpa_pv_gemm(
    const _Float16* __restrict__ probs, const _Float16* __restrict__ vhT,
    const float* __restrict__ psink, const float* __restrict__ v_null,
    float* __restrict__ out) {
  int b = blockIdx.z;
  int m0 = blockIdx.x * 128, n0 = blockIdx.y * 128;
  const _Float16* Ab = probs + (size_t)b * TT * TT;
  const _Float16* Bb = vhT + (size_t)b * TT;   // column offset into token axis
  __shared__ _Float16 As[CPA_TILE_ELEMS];
  __shared__ _Float16 Bs[CPA_TILE_ELEMS];
  v8f acc[2][4] = {};
  int kend = m0 + 128;  // probs beyond the causal diagonal are exactly 0
  cpa_mma_loop(Ab, TT, Bb, NROW, m0, n0, kend, As, Bs, acc);

  CPA_EPILOGUE_BEGIN
    size_t grow = (size_t)b * TT + row;
    out[grow * 512 + col] = val + psink[grow] * v_null[col];
  CPA_EPILOGUE_END
}

// ================================ host side ================================

extern "C" void kernel_launch(void* const* d_in, const int* in_sizes, int n_in,
                              void* d_out, int out_size, void* d_ws, size_t ws_size,
                              hipStream_t stream) {
  (void)in_sizes; (void)n_in; (void)out_size; (void)ws_size;
  const float* x      = (const float*)d_in[0];
  const float* Wq     = (const float*)d_in[1];
  const float* Wk     = (const float*)d_in[2];
  const float* Wv     = (const float*)d_in[3];
  const float* Wg     = (const float*)d_in[4];
  const float* bg     = (const float*)d_in[5];
  const float* Amat   = (const float*)d_in[6];
  const float* sink   = (const float*)d_in[7];
  const float* v_null = (const float*)d_in[8];
  float* out = (float*)d_out;

  char* ws = (char*)d_ws;
  size_t off = 0;
  auto take = [&](size_t bytes) -> size_t {
    size_t r = off; off += (bytes + 255) & ~(size_t)255; return r;
  };

  // persistent buffers
  _Float16* Wcat  = (_Float16*)(ws + take((size_t)2048 * 512 * 2));
  _Float16* ShT   = (_Float16*)(ws + take((size_t)512 * 512 * 2));
  _Float16* vhT   = (_Float16*)(ws + take((size_t)NROW * 512 * 2));
  _Float16* Qg    = (_Float16*)(ws + take((size_t)NROW * 512 * 2));
  _Float16* Kg    = (_Float16*)(ws + take((size_t)NROW * 512 * 2));
  _Float16* probs = (_Float16*)(ws + take((size_t)NROW * TT * 2));
  float*    psink = (float*)(ws + take((size_t)NROW * 4));

  // reusable region: attn (f32, 128MiB) overlays xh/q/k/v/g after they die
  size_t regionBase = off;
  _Float16* xh = (_Float16*)(ws + take((size_t)NROW * 512 * 2));
  float* qb = (float*)(ws + take((size_t)NROW * 512 * 4));
  float* kb = (float*)(ws + take((size_t)NROW * 512 * 4));
  float* vb = (float*)(ws + take((size_t)NROW * 512 * 4));
  float* gb = (float*)(ws + take((size_t)NROW * 512 * 4));
  float* Qc = (float*)(ws + take((size_t)NROW * 512 * 4));
  _Float16* Rq = (_Float16*)(ws + take((size_t)NROW * 512 * 2));
  _Float16* Rk = (_Float16*)(ws + take((size_t)NROW * 512 * 2));
  float* attnb = (float*)(ws + regionBase);  // NROW*TT*4 fits in xh..gb span

  // 1) converts
  { int n = NROW * 512;
    cpa_cast_f16<<<(n + 255) / 256, 256, 0, stream>>>(x, xh, n); }
  cpa_build_wcat<<<(2048 * 512 + 255) / 256, 256, 0, stream>>>(Wq, Wk, Wv, Wg, Wcat);
  cpa_build_st<<<(512 * 512 + 255) / 256, 256, 0, stream>>>(Amat, ShT);

  // 2) projections
  cpa_proj_gemm<<<dim3(NROW / 128, 2048 / 128), 256, 0, stream>>>(
      xh, Wcat, bg, qb, kb, vb, gb);

  // 3) serial chunk scan
  cpa_scan<<<BB, 512, 0, stream>>>(qb, kb, gb, Qc);

  // 4) norm + rope (+ transposed v cast)
  cpa_norm_rope<<<NROW, 512, 0, stream>>>(Qc, kb, vb, Rq, Rk, vhT);

  // 5) wedge
  cpa_wedge_gemm<<<dim3(NROW / 128, 512 / 128, 2), 256, 0, stream>>>(
      Rq, Rk, ShT, Qg, Kg);

  // 6) causal scores
  cpa_score_gemm<<<dim3(TT / 128, TT / 128, BB), 256, 0, stream>>>(Qg, Kg, attnb);

  // 7) softmax with sink
  cpa_softmax<<<NROW, 256, 0, stream>>>(attnb, sink, probs, psink);

  // 8) PV + sink * v_null
  cpa_pv_gemm<<<dim3(TT / 128, 512 / 128, BB), 256, 0, stream>>>(
      probs, vhT, psink, v_null, out);
}
